// VanillaEncoder_6176162972374
// MI455X (gfx1250) — matrix operand order repaired and checked by reading, not compiled, not run on hardware
//
#include <hip/hip_runtime.h>
#include <math.h>

// ---------------- problem constants ----------------
#define Bv   32
#define Nv   196
#define Cv   768
#define Hv   12
#define HDv  64
#define Lv   4
#define DFFv 3072
#define Mrows (Bv * Nv)      // 6272
#define C3   (3 * Cv)        // 2304
#define MT_ATTN 13           // ceil(196/16)
#define KPAD 224             // 196 padded to multiple of 32

typedef __attribute__((ext_vector_type(16))) _Float16 v16h;
typedef __attribute__((ext_vector_type(8)))  float    v8f;

union AFrag { v16h h; int4 q[2]; };

__device__ __forceinline__ v16h zero_frag16() {
    v16h z;
#pragma unroll
    for (int i = 0; i < 16; ++i) z[i] = (_Float16)0.f;
    return z;
}
__device__ __forceinline__ v8f zero_frag8() {
    v8f z;
#pragma unroll
    for (int i = 0; i < 8; ++i) z[i] = 0.f;
    return z;
}

// D = A*B + C, f16 inputs, f32 accum (V_WMMA_F32_16X16X32_F16)
__device__ __forceinline__ v8f wmma32(v16h a, v16h b, v8f c) {
    return __builtin_amdgcn_wmma_f32_16x16x32_f16(
        false, a, false, b, (short)0, c, false, false);
}

// A fragment (16x32 f16) from row-major A, contiguous-k lane layout used here (B fragments use the same k order):
//   lane = (kh<<4)|r ; halves 0..15 = A[r][k0+kh*16 .. +15]
__device__ __forceinline__ v16h load_afrag(const _Float16* base, int ld, int lane) {
    const int r = lane & 15, kh = lane >> 4;
    const _Float16* p = base + (size_t)r * ld + kh * 16;
    AFrag f;
    f.q[0] = *reinterpret_cast<const int4*>(p);
    f.q[1] = *reinterpret_cast<const int4*>(p + 8);
    return f.h;
}
// row-bounded version (rows beyond `rows` -> zeros)
__device__ __forceinline__ v16h load_afrag_rb(const _Float16* base, int ld, int lane, int rows) {
    const int r = lane & 15, kh = lane >> 4;
    if (r < rows) {
        const _Float16* p = base + (size_t)r * ld + kh * 16;
        AFrag f;
        f.q[0] = *reinterpret_cast<const int4*>(p);
        f.q[1] = *reinterpret_cast<const int4*>(p + 8);
        return f.h;
    }
    return zero_frag16();
}
// B fragment where lane's 16 halves are contiguous in memory (packed weights)
__device__ __forceinline__ v16h load_bfrag_contig(const _Float16* p) {
    AFrag f;
    f.q[0] = *reinterpret_cast<const int4*>(p);
    f.q[1] = *reinterpret_cast<const int4*>(p + 8);
    return f.h;
}
// B fragment for S = Q*K^T : B[kk][n] = Kmat[n][kk] -> contiguous along Kmat row n
__device__ __forceinline__ v16h load_bfrag_kT(const _Float16* kbase, int ld, int k0,
                                              int lane, int rows) {
    const int n = lane & 15, kh = lane >> 4;
    if (n < rows) {
        const _Float16* p = kbase + (size_t)n * ld + k0 + kh * 16;
        AFrag f;
        f.q[0] = *reinterpret_cast<const int4*>(p);
        f.q[1] = *reinterpret_cast<const int4*>(p + 8);
        return f.h;
    }
    return zero_frag16();
}
// B fragment for O = P*V : B[kk][n] = V[kk][n], K-major -> scalar gathers w/ row bound
__device__ __forceinline__ v16h load_bfrag_v(const _Float16* vbase, int ld, int k0,
                                             int lane, int rows) {
    const int n = lane & 15, kh = lane >> 4;
    v16h b;
#pragma unroll
    for (int i = 0; i < 16; ++i) {
        const int kk = k0 + kh * 16 + i;
        b[i] = (kk < rows) ? vbase[(size_t)kk * ld + n] : (_Float16)0.f;
    }
    return b;
}

// ---------------- block reductions (blockDim = 256) ----------------
__device__ __forceinline__ float block_sum(float v, float* red) {
    const int t = threadIdx.x;
    red[t] = v; __syncthreads();
    for (int s = 128; s > 0; s >>= 1) {
        if (t < s) red[t] += red[t + s];
        __syncthreads();
    }
    float r = red[0]; __syncthreads();
    return r;
}
__device__ __forceinline__ float block_max(float v, float* red) {
    const int t = threadIdx.x;
    red[t] = v; __syncthreads();
    for (int s = 128; s > 0; s >>= 1) {
        if (t < s) red[t] = fmaxf(red[t], red[t + s]);
        __syncthreads();
    }
    float r = red[0]; __syncthreads();
    return r;
}

// ---------------- latent bottleneck fusion ----------------
// fused[b,l,:] = softmax_k( lat_l . concat_k * C^-0.5 ) @ concat   (one block / batch)
__global__ __launch_bounds__(256) void fuse_kernel(const float* __restrict__ x,
                                                   const float* __restrict__ y,
                                                   const float* __restrict__ lat,
                                                   float* __restrict__ fused) {
    __shared__ float latS[Lv * Cv];
    __shared__ float sc[Lv * (2 * Nv)];
    __shared__ float red[256];
    const int b = blockIdx.x, t = threadIdx.x;
    for (int i = t; i < Lv * Cv; i += 256) latS[i] = lat[i];
    __syncthreads();
    const float scale = 0.036084391824f;   // 768^-0.5
    for (int k = t; k < 2 * Nv; k += 256) {
        const float* row = (k < Nv) ? x + ((size_t)b * Nv + k) * Cv
                                    : y + ((size_t)b * Nv + (k - Nv)) * Cv;
        float d0 = 0.f, d1 = 0.f, d2 = 0.f, d3 = 0.f;
        for (int c = 0; c < Cv; ++c) {
            const float v = row[c];
            d0 += v * latS[c];
            d1 += v * latS[Cv + c];
            d2 += v * latS[2 * Cv + c];
            d3 += v * latS[3 * Cv + c];
        }
        sc[0 * (2 * Nv) + k] = d0 * scale;
        sc[1 * (2 * Nv) + k] = d1 * scale;
        sc[2 * (2 * Nv) + k] = d2 * scale;
        sc[3 * (2 * Nv) + k] = d3 * scale;
    }
    __syncthreads();
    for (int l = 0; l < Lv; ++l) {
        float m = -3.0e38f;
        for (int k = t; k < 2 * Nv; k += 256) m = fmaxf(m, sc[l * (2 * Nv) + k]);
        m = block_max(m, red);
        float s = 0.f;
        for (int k = t; k < 2 * Nv; k += 256) {
            const float e = expf(sc[l * (2 * Nv) + k] - m);
            sc[l * (2 * Nv) + k] = e;
            s += e;
        }
        s = block_sum(s, red);
        const float inv = 1.f / s;
        for (int k = t; k < 2 * Nv; k += 256) sc[l * (2 * Nv) + k] *= inv;
        __syncthreads();
    }
    for (int c = t; c < Cv; c += 256) {
        float o0 = 0.f, o1 = 0.f, o2 = 0.f, o3 = 0.f;
        for (int k = 0; k < 2 * Nv; ++k) {
            const float v = (k < Nv) ? x[((size_t)b * Nv + k) * Cv + c]
                                     : y[((size_t)b * Nv + (k - Nv)) * Cv + c];
            o0 += sc[0 * (2 * Nv) + k] * v;
            o1 += sc[1 * (2 * Nv) + k] * v;
            o2 += sc[2 * (2 * Nv) + k] * v;
            o3 += sc[3 * (2 * Nv) + k] * v;
        }
        float* fb = fused + (size_t)b * Lv * Cv;
        fb[c] = o0; fb[Cv + c] = o1; fb[2 * Cv + c] = o2; fb[3 * Cv + c] = o3;
    }
}

// out_row = x_row + scale * softmax(x_row . fused_l * C^-0.5) @ fused  (block / row)
__global__ __launch_bounds__(256) void cross_attn_kernel(const float* __restrict__ X,
                                                         const float* __restrict__ fused,
                                                         const float* __restrict__ scale_p,
                                                         float* __restrict__ out) {
    __shared__ float red[256];
    __shared__ float p[Lv];
    const int row = blockIdx.x, t = threadIdx.x;
    const int b = row / Nv;
    const float* xr = X + (size_t)row * Cv;
    const float* fb = fused + (size_t)b * Lv * Cv;
    float d0 = 0.f, d1 = 0.f, d2 = 0.f, d3 = 0.f;
    for (int c = t; c < Cv; c += 256) {
        const float v = xr[c];
        d0 += v * fb[c]; d1 += v * fb[Cv + c];
        d2 += v * fb[2 * Cv + c]; d3 += v * fb[3 * Cv + c];
    }
    const float scale = 0.036084391824f;
    float r0 = block_sum(d0, red); if (t == 0) p[0] = r0 * scale;
    float r1 = block_sum(d1, red); if (t == 0) p[1] = r1 * scale;
    float r2 = block_sum(d2, red); if (t == 0) p[2] = r2 * scale;
    float r3 = block_sum(d3, red); if (t == 0) p[3] = r3 * scale;
    __syncthreads();
    if (t == 0) {
        const float mx = fmaxf(fmaxf(p[0], p[1]), fmaxf(p[2], p[3]));
        const float e0 = expf(p[0] - mx), e1 = expf(p[1] - mx);
        const float e2 = expf(p[2] - mx), e3 = expf(p[3] - mx);
        const float inv = 1.f / (e0 + e1 + e2 + e3);
        p[0] = e0 * inv; p[1] = e1 * inv; p[2] = e2 * inv; p[3] = e3 * inv;
    }
    __syncthreads();
    const float sa = scale_p[0];
    const float p0 = p[0], p1 = p[1], p2 = p[2], p3 = p[3];
    for (int c = t; c < Cv; c += 256) {
        const float o = p0 * fb[c] + p1 * fb[Cv + c] + p2 * fb[2 * Cv + c] + p3 * fb[3 * Cv + c];
        out[(size_t)row * Cv + c] = xr[c] + sa * o;
    }
}

// ---------------- layernorm -> f16 (block / row) ----------------
__global__ __launch_bounds__(256) void ln_kernel(const float* __restrict__ X,
                                                 const float* __restrict__ g,
                                                 const float* __restrict__ bb,
                                                 _Float16* __restrict__ Hh) {
    __shared__ float red[256];
    const int row = blockIdx.x, t = threadIdx.x;
    const float* xr = X + (size_t)row * Cv;
    float s = 0.f;
    for (int c = t; c < Cv; c += 256) s += xr[c];
    const float mean = block_sum(s, red) * (1.f / Cv);
    float vs = 0.f;
    for (int c = t; c < Cv; c += 256) { const float d = xr[c] - mean; vs += d * d; }
    const float var = block_sum(vs, red) * (1.f / Cv);
    const float inv = rsqrtf(var + 1e-6f);
    for (int c = t; c < Cv; c += 256)
        Hh[(size_t)row * Cv + c] = (_Float16)((xr[c] - mean) * inv * g[c] + bb[c]);
}

// ---------------- pack W[K,N] f32 -> f16 B-fragment-major ----------------
// dst[((nt*(K/32)+kt)*32 + lane)*16 + i] = W[kt*32 + (lane>>4)*16 + i][nt*16 + (lane&15)]
__global__ __launch_bounds__(256) void pack_w_kernel(const float* __restrict__ W,
                                                     _Float16* __restrict__ Wp,
                                                     int K, int N) {
    const int idx = blockIdx.x * 256 + threadIdx.x;
    if (idx >= K * N) return;
    const int i    = idx & 15;
    const int lane = (idx >> 4) & 31;
    const int tile = idx >> 9;
    const int kSteps = K >> 5;
    const int kt = tile % kSteps;
    const int nt = tile / kSteps;
    const int kh = lane >> 4, n = lane & 15;
    const int k   = kt * 32 + kh * 16 + i;
    const int col = nt * 16 + n;
    Wp[idx] = (_Float16)W[(size_t)k * N + col];
}

// ---------------- WMMA GEMM: out = A[M,K](f16) @ Wp + bias ----------------
// EPI 0: store f16 ; 1: exact GELU -> f16 ; 2: residual f32 += (in d_out)
// wave tile = 64x64 (16 wmma / k-step; each A/B fragment reused 4x).
// M%64==0, N%64==0, K%32==0 guaranteed for all call sites.
template <int EPI>
__global__ __launch_bounds__(128) void gemm_wmma(const _Float16* __restrict__ A,
                                                 const _Float16* __restrict__ Wp,
                                                 const float* __restrict__ bias,
                                                 float* __restrict__ resid,
                                                 _Float16* __restrict__ out16,
                                                 int Mm, int Nn, int Kk) {
    const int lane = threadIdx.x & 31;
    const int wid  = blockIdx.x * (blockDim.x >> 5) + (threadIdx.x >> 5);
    const int nTiles = Nn >> 6;                 // 64-wide N tiles
    const int total  = (Mm >> 6) * nTiles;
    if (wid >= total) return;
    const int mt = wid / nTiles, nt = wid % nTiles;
    const int m0 = mt << 6, n0 = nt << 6;
    const int kSteps = Kk >> 5;

    v8f acc[4][4];
#pragma unroll
    for (int i = 0; i < 4; ++i)
#pragma unroll
        for (int j = 0; j < 4; ++j) acc[i][j] = zero_frag8();

    const _Float16* Abase = A + (size_t)m0 * Kk;
    // per-lane packed-B stream: chunk(j, ks) = Bbase + (j*kSteps + ks)*512
    const _Float16* Bbase = Wp + (size_t)(n0 >> 4) * kSteps * 512 + lane * 16;

    for (int ks = 0; ks < kSteps; ++ks) {
        const int k0 = ks << 5;
        if (ks + 1 < kSteps) {   // prefetch next K-step panels (global_prefetch_b8)
            __builtin_prefetch(Abase + k0 + 32, 0, 1);
            __builtin_prefetch(Bbase + (size_t)(ks + 1) * 512, 0, 1);
        }
        v16h a0 = load_afrag(Abase + k0, Kk, lane);
        v16h a1 = load_afrag(Abase + (size_t)16 * Kk + k0, Kk, lane);
        v16h a2 = load_afrag(Abase + (size_t)32 * Kk + k0, Kk, lane);
        v16h a3 = load_afrag(Abase + (size_t)48 * Kk + k0, Kk, lane);
        v16h b0 = load_bfrag_contig(Bbase + ((size_t)0 * kSteps + ks) * 512);
        v16h b1 = load_bfrag_contig(Bbase + ((size_t)1 * kSteps + ks) * 512);
        v16h b2 = load_bfrag_contig(Bbase + ((size_t)2 * kSteps + ks) * 512);
        v16h b3 = load_bfrag_contig(Bbase + ((size_t)3 * kSteps + ks) * 512);
        acc[0][0] = wmma32(a0, b0, acc[0][0]);
        acc[1][0] = wmma32(a1, b0, acc[1][0]);
        acc[2][0] = wmma32(a2, b0, acc[2][0]);
        acc[3][0] = wmma32(a3, b0, acc[3][0]);
        acc[0][1] = wmma32(a0, b1, acc[0][1]);
        acc[1][1] = wmma32(a1, b1, acc[1][1]);
        acc[2][1] = wmma32(a2, b1, acc[2][1]);
        acc[3][1] = wmma32(a3, b1, acc[3][1]);
        acc[0][2] = wmma32(a0, b2, acc[0][2]);
        acc[1][2] = wmma32(a1, b2, acc[1][2]);
        acc[2][2] = wmma32(a2, b2, acc[2][2]);
        acc[3][2] = wmma32(a3, b2, acc[3][2]);
        acc[0][3] = wmma32(a0, b3, acc[0][3]);
        acc[1][3] = wmma32(a1, b3, acc[1][3]);
        acc[2][3] = wmma32(a2, b3, acc[2][3]);
        acc[3][3] = wmma32(a3, b3, acc[3][3]);
    }

    const int col  = lane & 15;
    const int rsel = (lane >> 4) << 3;   // lanes 16-31 hold rows +8
#pragma unroll
    for (int j = 0; j < 4; ++j) {
        const int c = n0 + j * 16 + col;
        const float bvv = bias[c];
#pragma unroll
        for (int i = 0; i < 4; ++i) {
            const int mrow = m0 + i * 16 + rsel;
#pragma unroll
            for (int v = 0; v < 8; ++v) {
                const int row = mrow + v;
                const float val = acc[i][j][v] + bvv;
                const size_t idx = (size_t)row * Nn + c;
                if (EPI == 0) {
                    out16[idx] = (_Float16)val;
                } else if (EPI == 1) {
                    out16[idx] = (_Float16)(0.5f * val * (1.f + erff(val * 0.7071067811865475f)));
                } else {
                    resid[idx] = resid[idx] + val;
                }
            }
        }
    }
}

// ---------------- fused per-head attention (one wave / (b,h,mtile)) ----------
// qkv rows: [b*196+n] x 2304 f16 ; q @ +0, k @ +768, v @ +1536 (per head h*64)
__global__ __launch_bounds__(32) void attn_kernel(const _Float16* __restrict__ qkv,
                                                  _Float16* __restrict__ aout) {
    __shared__ __align__(16) float     S[16][KPAD];
    __shared__ __align__(16) _Float16  P[16][KPAD];
    const int lane = threadIdx.x;
    int bid = blockIdx.x;
    const int mt = bid % MT_ATTN; bid /= MT_ATTN;
    const int h  = bid % Hv;      const int b = bid / Hv;

    const _Float16* qb = qkv + ((size_t)b * Nv) * C3 + h * HDv;
    const _Float16* kb = qb + Cv;
    const _Float16* vb = qb + 2 * Cv;
    const int m0 = mt * 16;
    int qrows = Nv - m0; if (qrows > 16) qrows = 16;

    // S = q @ k^T  (K = 64 -> two 32-steps)
    v16h aq0 = load_afrag_rb(qb + (size_t)m0 * C3 + 0,  C3, lane, qrows);
    v16h aq1 = load_afrag_rb(qb + (size_t)m0 * C3 + 32, C3, lane, qrows);
    for (int nt = 0; nt < MT_ATTN; ++nt) {
        int krows = Nv - nt * 16; if (krows > 16) krows = 16;
        const _Float16* kt = kb + (size_t)nt * 16 * C3;
        v8f acc = zero_frag8();
        acc = wmma32(aq0, load_bfrag_kT(kt, C3, 0,  lane, krows), acc);
        acc = wmma32(aq1, load_bfrag_kT(kt, C3, 32, lane, krows), acc);
        const int cc = nt * 16 + (lane & 15);
        const int rb0 = (lane >> 4) << 3;
#pragma unroll
        for (int v = 0; v < 8; ++v) S[v + rb0][cc] = acc[v];
    }
    __syncthreads();

    // f32 softmax over the 196 valid keys (scale = 64^-0.5 = 0.125)
    if (lane < 16) {
        float mx = -3.0e38f;
        for (int c = 0; c < Nv; ++c) mx = fmaxf(mx, S[lane][c] * 0.125f);
        float sum = 0.f;
        for (int c = 0; c < Nv; ++c) {
            const float e = expf(S[lane][c] * 0.125f - mx);
            S[lane][c] = e;
            sum += e;
        }
        const float inv = 1.f / sum;
        for (int c = 0; c < Nv; ++c)    P[lane][c] = (_Float16)(S[lane][c] * inv);
        for (int c = Nv; c < KPAD; ++c) P[lane][c] = (_Float16)0.f;
    }
    __syncthreads();

    // O = P @ V  (K padded to 224)
    v8f o0 = zero_frag8(), o1 = zero_frag8(), o2 = zero_frag8(), o3 = zero_frag8();
    for (int ks = 0; ks < KPAD / 32; ++ks) {
        const int k0 = ks * 32;
        v16h pa = load_afrag(&P[0][0] + k0, KPAD, lane);
        o0 = wmma32(pa, load_bfrag_v(vb + 0,  C3, k0, lane, Nv), o0);
        o1 = wmma32(pa, load_bfrag_v(vb + 16, C3, k0, lane, Nv), o1);
        o2 = wmma32(pa, load_bfrag_v(vb + 32, C3, k0, lane, Nv), o2);
        o3 = wmma32(pa, load_bfrag_v(vb + 48, C3, k0, lane, Nv), o3);
    }
    const int col = lane & 15;
    const int rb  = (lane >> 4) << 3;
    _Float16* obase = aout + ((size_t)b * Nv + m0) * Cv + (size_t)h * HDv + col;
#pragma unroll
    for (int v = 0; v < 8; ++v) {
        const int row = m0 + v + rb;
        if (row < Nv) {
            _Float16* orow = obase + (size_t)(v + rb) * Cv;
            orow[0]  = (_Float16)o0[v];
            orow[16] = (_Float16)o1[v];
            orow[32] = (_Float16)o2[v];
            orow[48] = (_Float16)o3[v];
        }
    }
}

// ---------------- host-side launch ----------------
static inline int gemm_blocks(int Mm, int Nn) {
    const int waves = (Mm / 64) * (Nn / 64);
    return (waves + 3) / 4;
}

extern "C" void kernel_launch(void* const* d_in, const int* in_sizes, int n_in,
                              void* d_out, int out_size, void* d_ws, size_t ws_size,
                              hipStream_t stream) {
    (void)in_sizes; (void)n_in; (void)out_size; (void)ws_size;
    const float* x       = (const float*)d_in[0];
    const float* y       = (const float*)d_in[1];
    const float* lat     = (const float*)d_in[2];
    const float* scale_a = (const float*)d_in[3];
    const float* scale_v = (const float*)d_in[4];

    float* out_x = (float*)d_out;
    float* out_y = out_x + (size_t)Mrows * Cv;

    // workspace carve (all chunk sizes are multiples of 256 bytes)
    char* w = (char*)d_ws;
    float*    fused  = (float*)w;    w += (size_t)Bv * Lv * Cv * 4;       // 384 KB
    _Float16* hA     = (_Float16*)w; w += (size_t)Mrows * Cv * 2;         // 9.6 MB
    _Float16* big    = (_Float16*)w; w += (size_t)Mrows * DFFv * 2;       // 38.5 MB (qkv16 / fc1-out)
    _Float16* attn16 = (_Float16*)w; w += (size_t)Mrows * Cv * 2;         // 9.6 MB
    _Float16* wpack  = (_Float16*)w;                                      // 4.7 MB

    // latent fusion + gated cross-attention (writes f32 residual stream into d_out)
    fuse_kernel<<<Bv, 256, 0, stream>>>(x, y, lat, fused);
    cross_attn_kernel<<<Mrows, 256, 0, stream>>>(x, fused, scale_a, out_x);
    cross_attn_kernel<<<Mrows, 256, 0, stream>>>(y, fused, scale_v, out_y);

    for (int s2 = 0; s2 < 2; ++s2) {
        const int base = 5 + s2 * 12;
        const float* ln1_g  = (const float*)d_in[base + 0];
        const float* ln1_b  = (const float*)d_in[base + 1];
        const float* qkv_w  = (const float*)d_in[base + 2];
        const float* qkv_b  = (const float*)d_in[base + 3];
        const float* proj_w = (const float*)d_in[base + 4];
        const float* proj_b = (const float*)d_in[base + 5];
        const float* ln2_g  = (const float*)d_in[base + 6];
        const float* ln2_b  = (const float*)d_in[base + 7];
        const float* fc1_w  = (const float*)d_in[base + 8];
        const float* fc1_b  = (const float*)d_in[base + 9];
        const float* fc2_w  = (const float*)d_in[base + 10];
        const float* fc2_b  = (const float*)d_in[base + 11];
        float* xo = s2 ? out_y : out_x;

        // --- attention block ---
        ln_kernel<<<Mrows, 256, 0, stream>>>(xo, ln1_g, ln1_b, hA);
        pack_w_kernel<<<(Cv * C3 + 255) / 256, 256, 0, stream>>>(qkv_w, wpack, Cv, C3);
        gemm_wmma<0><<<gemm_blocks(Mrows, C3), 128, 0, stream>>>(
            hA, wpack, qkv_b, nullptr, big, Mrows, C3, Cv);
        attn_kernel<<<Bv * Hv * MT_ATTN, 32, 0, stream>>>(big, attn16);
        pack_w_kernel<<<(Cv * Cv + 255) / 256, 256, 0, stream>>>(proj_w, wpack, Cv, Cv);
        gemm_wmma<2><<<gemm_blocks(Mrows, Cv), 128, 0, stream>>>(
            attn16, wpack, proj_b, xo, nullptr, Mrows, Cv, Cv);

        // --- MLP block ---
        ln_kernel<<<Mrows, 256, 0, stream>>>(xo, ln2_g, ln2_b, hA);
        pack_w_kernel<<<(Cv * DFFv + 255) / 256, 256, 0, stream>>>(fc1_w, wpack, Cv, DFFv);
        gemm_wmma<1><<<gemm_blocks(Mrows, DFFv), 128, 0, stream>>>(
            hA, wpack, fc1_b, nullptr, big, Mrows, DFFv, Cv);
        pack_w_kernel<<<(DFFv * Cv + 255) / 256, 256, 0, stream>>>(fc2_w, wpack, DFFv, Cv);
        gemm_wmma<2><<<gemm_blocks(Mrows, Cv), 128, 0, stream>>>(
            big, wpack, fc2_b, xo, nullptr, Mrows, Cv, DFFv);
    }
}
